// UOTHead_33818572489500
// MI455X (gfx1250) — matrix operand order repaired and checked by reading, not compiled, not run on hardware
//
#include <hip/hip_runtime.h>
#include <hip/hip_bf16.h>

typedef __attribute__((ext_vector_type(16))) _Float16 v16h;
typedef __attribute__((ext_vector_type(8)))  _Float16 v8h;
typedef __attribute__((ext_vector_type(8)))  float    v8f;

#define NPTS   2048
#define NFEAT  256
#define NBATCH 8
#define NTILES (NPTS / 16)     // 128 16-wide tiles
#define SLABW  64              // output columns/rows per block
#define NSLABS (NPTS / SLABW)  // 32 blocks along the output dim
#define HTILES (NTILES / 2)    // 64 tiles per reduction half

// ---------------------------------------------------------------------------
// Fragment loaders for V_WMMA_F32_16X16X32_F16 (contiguous k order, wave32)
// A-matrix 16x32 f16: lane&15 = row M. lanes 0-15: K=0..15, lanes 16-31:
// K=16..31 (contiguous, matching B).  -> two 16B loads per k-step.
// ---------------------------------------------------------------------------
__device__ __forceinline__ v16h load_a_frag(const _Float16* __restrict__ row,
                                            int k0, int hi) {
  const v8h* p0 = reinterpret_cast<const v8h*>(row + k0 + hi * 16);
  const v8h* p1 = reinterpret_cast<const v8h*>(row + k0 + 8 + hi * 16);
  v8h x = *p0, y = *p1;
  v16h r;
#pragma unroll
  for (int i = 0; i < 8; ++i) { r[i] = x[i]; r[i + 8] = y[i]; }
  return r;
}

// B-matrix 32x16 f16 (stored as B^T rows, i.e. f2h[m][c] row-major):
// lane&15 = column N; VGPR r = rows K=2r,2r+1 (+16 for hi lanes)
// -> one contiguous 32B load per k-step.
__device__ __forceinline__ v16h load_b_frag(const _Float16* __restrict__ rowT,
                                            int k0, int hi) {
  return *reinterpret_cast<const v16h*>(rowT + k0 + hi * 16);
}

// ---------------------------------------------------------------------------
// 1) L2-normalize features, transpose (B2,C,N)->(B2,N,C), convert to f16.
// ---------------------------------------------------------------------------
__global__ void k_normalize(const float* __restrict__ feats,
                            _Float16* __restrict__ fh) {
  const int n  = blockIdx.x * blockDim.x + threadIdx.x;  // 0..2047
  const int b2 = blockIdx.y;                             // 0..15
  const float* src = feats + (size_t)b2 * NFEAT * NPTS;
  float ss = 0.f;
#pragma unroll 4
  for (int c = 0; c < NFEAT; ++c) {
    float v = src[(size_t)c * NPTS + n];
    ss += v * v;
  }
  const float rn = rsqrtf(ss + 1e-8f);
  _Float16* dst = fh + ((size_t)b2 * NPTS + n) * NFEAT;
#pragma unroll 4
  for (int c = 0; c < NFEAT; ++c)
    dst[c] = (_Float16)(src[(size_t)c * NPTS + n] * rn);
}

__global__ void k_init_a(float* __restrict__ avec) {
  avec[blockIdx.x * blockDim.x + threadIdx.x] = 1.0f / (float)NPTS;
}

// ---------------------------------------------------------------------------
// 2) KTa[m] = sum_n K[n,m] * a[n];  b = (prob2/(KTa+1e-8))^power
//    Block owns a 64-wide m-slab. Wave w: slab = w&3 (16 columns),
//    nhalf = w>>2 (which half of the n sweep). The 4 slab-owner waves of
//    each half sweep identical f1 tiles in lockstep -> WGP$ serves 3/4 of
//    the streamed reads (4x L2 traffic cut vs. 16-wide blocks).
// ---------------------------------------------------------------------------
__global__ void k_colsum_b(const _Float16* __restrict__ fh,
                           const float* __restrict__ avec,
                           float* __restrict__ bvec,
                           const float* __restrict__ eps_p,
                           const float* __restrict__ gam_p) {
  const int bb = blockIdx.y;
  const int tid = threadIdx.x, wave = tid >> 5, lane = tid & 31;
  const int ml = lane & 15, hi = lane >> 4;
  const int slab = wave & 3, nhalf = wave >> 2;
  const float eps = __expf(eps_p[0]) + 0.03f;
  const float gam = __expf(gam_p[0]);
  const float power = gam / (gam + eps + 1e-8f);
  const float inv_eps = 1.0f / eps;
  const _Float16* f1 = fh + (size_t)bb * NPTS * NFEAT;
  const _Float16* f2 = fh + (size_t)(NBATCH + bb) * NPTS * NFEAT;
  const float* a = avec + (size_t)bb * NPTS;

  __shared__ float acc_m[SLABW];
  if (tid < SLABW) acc_m[tid] = 0.f;
  __syncthreads();

  const int m0 = blockIdx.x * SLABW + slab * 16;
  const _Float16* brow = f2 + (size_t)(m0 + ml) * NFEAT;
  v16h bf[8];
#pragma unroll
  for (int kk = 0; kk < 8; ++kk) bf[kk] = load_b_frag(brow, kk * 32, hi);

  float part = 0.f;
  for (int t = nhalf * HTILES; t < (nhalf + 1) * HTILES; ++t) {
    const int n0 = t * 16;
    const _Float16* arow = f1 + (size_t)(n0 + ml) * NFEAT;
    v8f d = {};
#pragma unroll
    for (int kk = 0; kk < 8; ++kk) {
      v16h af = load_a_frag(arow, kk * 32, hi);
      d = __builtin_amdgcn_wmma_f32_16x16x32_f16(false, af, false, bf[kk],
                                                 (short)0, d, false, false);
    }
#pragma unroll
    for (int r = 0; r < 8; ++r) {
      float kv = __expf((d[r] - 1.0f) * inv_eps);       // K = exp((dot-1)/eps)
      part += kv * a[n0 + r + hi * 8];
    }
  }
  // lane & lane+16 hold complementary row-halves of the same column
  part += __shfl_xor(part, 16);
  if (hi == 0) atomicAdd(&acc_m[slab * 16 + ml], part);  // combine two n-halves
  __syncthreads();
  if (tid < SLABW) {
    const int m = blockIdx.x * SLABW + tid;
    float kta  = acc_m[tid];
    float bnew = __powf((1.0f / (float)NPTS) / (kta + 1e-8f), power);
    bvec[(size_t)bb * NPTS + m] = bnew;
  }
}

// ---------------------------------------------------------------------------
// 3) Kb[n] = sum_m K[n,m]*b[m];  a = (prob1/(Kb+1e-8))^power
//    Block owns a 64-row n-slab; wave: slab = w&3, mhalf = w>>2.
//    A fragments loop-invariant -> preloaded; streamed f2 tiles shared
//    across the 4 slab-owner waves via WGP$.
// ---------------------------------------------------------------------------
__global__ void k_rowsum_a(const _Float16* __restrict__ fh,
                           const float* __restrict__ bvec,
                           float* __restrict__ avec,
                           const float* __restrict__ eps_p,
                           const float* __restrict__ gam_p) {
  const int bb = blockIdx.y;
  const int tid = threadIdx.x, wave = tid >> 5, lane = tid & 31;
  const int ml = lane & 15, hi = lane >> 4;
  const int slab = wave & 3, mhalf = wave >> 2;
  const float eps = __expf(eps_p[0]) + 0.03f;
  const float gam = __expf(gam_p[0]);
  const float power = gam / (gam + eps + 1e-8f);
  const float inv_eps = 1.0f / eps;
  const _Float16* f1 = fh + (size_t)bb * NPTS * NFEAT;
  const _Float16* f2 = fh + (size_t)(NBATCH + bb) * NPTS * NFEAT;
  const float* bv = bvec + (size_t)bb * NPTS;

  __shared__ float acc_n[SLABW];
  if (tid < SLABW) acc_n[tid] = 0.f;
  __syncthreads();

  const int n0 = blockIdx.x * SLABW + slab * 16;
  const _Float16* arow = f1 + (size_t)(n0 + ml) * NFEAT;
  v16h af[8];
#pragma unroll
  for (int kk = 0; kk < 8; ++kk) af[kk] = load_a_frag(arow, kk * 32, hi);

  float rp[8];
#pragma unroll
  for (int r = 0; r < 8; ++r) rp[r] = 0.f;

  for (int t = mhalf * HTILES; t < (mhalf + 1) * HTILES; ++t) {
    const int m0 = t * 16;
    const _Float16* brow = f2 + (size_t)(m0 + ml) * NFEAT;
    v8f d = {};
#pragma unroll
    for (int kk = 0; kk < 8; ++kk) {
      v16h bfk = load_b_frag(brow, kk * 32, hi);
      d = __builtin_amdgcn_wmma_f32_16x16x32_f16(false, af[kk], false, bfk,
                                                 (short)0, d, false, false);
    }
    const float bm = bv[m0 + ml];
#pragma unroll
    for (int r = 0; r < 8; ++r)
      rp[r] += __expf((d[r] - 1.0f) * inv_eps) * bm;
  }
  // sum over the 16 lanes of each half (row n0 + r + 8*hi)
#pragma unroll
  for (int r = 0; r < 8; ++r) {
    float v = rp[r];
    v += __shfl_xor(v, 1); v += __shfl_xor(v, 2);
    v += __shfl_xor(v, 4); v += __shfl_xor(v, 8);
    if (ml == 0) atomicAdd(&acc_n[slab * 16 + r + hi * 8], v);
  }
  __syncthreads();
  if (tid < SLABW) {
    const int n = blockIdx.x * SLABW + tid;
    float kb   = acc_n[tid];
    float anew = __powf((1.0f / (float)NPTS) / (kb + 1e-8f), power);
    avec[(size_t)bb * NPTS + n] = anew;
  }
}

// ---------------------------------------------------------------------------
// 4) Final T statistics per row n (T = a K b^T):
//    corr[n] = sum_m T[n,m];  proj[n] = (T @ coords2)[n] / (corr+1e-8)
// ---------------------------------------------------------------------------
__global__ void k_final_stats(const _Float16* __restrict__ fh,
                              const float* __restrict__ avec,
                              const float* __restrict__ bvec,
                              const float* __restrict__ kpts,
                              float* __restrict__ corr,
                              float* __restrict__ proj,
                              const float* __restrict__ eps_p) {
  const int bb = blockIdx.y;
  const int tid = threadIdx.x, wave = tid >> 5, lane = tid & 31;
  const int ml = lane & 15, hi = lane >> 4;
  const int slab = wave & 3, mhalf = wave >> 2;
  const float eps = __expf(eps_p[0]) + 0.03f;
  const float inv_eps = 1.0f / eps;
  const _Float16* f1 = fh + (size_t)bb * NPTS * NFEAT;
  const _Float16* f2 = fh + (size_t)(NBATCH + bb) * NPTS * NFEAT;
  const float* bv = bvec + (size_t)bb * NPTS;
  const float* c2 = kpts + (size_t)(NBATCH + bb) * NPTS * 3;

  __shared__ float rs[SLABW], rx[SLABW], ry[SLABW], rz[SLABW];
  if (tid < SLABW) { rs[tid] = 0.f; rx[tid] = 0.f; ry[tid] = 0.f; rz[tid] = 0.f; }
  __syncthreads();

  const int n0 = blockIdx.x * SLABW + slab * 16;
  const _Float16* arow = f1 + (size_t)(n0 + ml) * NFEAT;
  v16h af[8];
#pragma unroll
  for (int kk = 0; kk < 8; ++kk) af[kk] = load_a_frag(arow, kk * 32, hi);

  float s[8], sx[8], sy[8], sz[8];
#pragma unroll
  for (int r = 0; r < 8; ++r) { s[r] = 0.f; sx[r] = 0.f; sy[r] = 0.f; sz[r] = 0.f; }

  for (int t = mhalf * HTILES; t < (mhalf + 1) * HTILES; ++t) {
    const int m0 = t * 16;
    const _Float16* brow = f2 + (size_t)(m0 + ml) * NFEAT;
    v8f d = {};
#pragma unroll
    for (int kk = 0; kk < 8; ++kk) {
      v16h bfk = load_b_frag(brow, kk * 32, hi);
      d = __builtin_amdgcn_wmma_f32_16x16x32_f16(false, af[kk], false, bfk,
                                                 (short)0, d, false, false);
    }
    const int m = m0 + ml;
    const float bm = bv[m];
    const float cx = c2[m * 3 + 0], cy = c2[m * 3 + 1], cz = c2[m * 3 + 2];
#pragma unroll
    for (int r = 0; r < 8; ++r) {
      float w = __expf((d[r] - 1.0f) * inv_eps) * bm;   // K[n,m]*b[m]
      s[r] += w; sx[r] += w * cx; sy[r] += w * cy; sz[r] += w * cz;
    }
  }
#pragma unroll
  for (int r = 0; r < 8; ++r) {
    const int nl = slab * 16 + r + hi * 8;
    float v0 = s[r],  v1 = sx[r], v2 = sy[r], v3 = sz[r];
    v0 += __shfl_xor(v0, 1); v0 += __shfl_xor(v0, 2); v0 += __shfl_xor(v0, 4); v0 += __shfl_xor(v0, 8);
    v1 += __shfl_xor(v1, 1); v1 += __shfl_xor(v1, 2); v1 += __shfl_xor(v1, 4); v1 += __shfl_xor(v1, 8);
    v2 += __shfl_xor(v2, 1); v2 += __shfl_xor(v2, 2); v2 += __shfl_xor(v2, 4); v2 += __shfl_xor(v2, 8);
    v3 += __shfl_xor(v3, 1); v3 += __shfl_xor(v3, 2); v3 += __shfl_xor(v3, 4); v3 += __shfl_xor(v3, 8);
    if (ml == 0) {
      atomicAdd(&rs[nl], v0); atomicAdd(&rx[nl], v1);
      atomicAdd(&ry[nl], v2); atomicAdd(&rz[nl], v3);
    }
  }
  __syncthreads();
  if (tid < SLABW) {
    const int n = blockIdx.x * SLABW + tid;
    const float aw = avec[(size_t)bb * NPTS + n];
    const float c  = aw * rs[tid];            // corr_sum[n]
    const float den = c + 1e-8f;
    corr[(size_t)bb * NPTS + n] = c;
    float* p = proj + ((size_t)bb * NPTS + n) * 3;
    p[0] = aw * rx[tid] / den;
    p[1] = aw * ry[tid] / den;
    p[2] = aw * rz[tid] / den;
  }
}

// ---------------------------------------------------------------------------
// 5) Per-batch weighted Kabsch with 3x3 Jacobi SVD.
// ---------------------------------------------------------------------------
__device__ __forceinline__ float det3(const float R[3][3]) {
  return R[0][0] * (R[1][1] * R[2][2] - R[1][2] * R[2][1])
       - R[0][1] * (R[1][0] * R[2][2] - R[1][2] * R[2][0])
       + R[0][2] * (R[1][0] * R[2][1] - R[1][1] * R[2][0]);
}

__global__ void k_transform(const float* __restrict__ corr,
                            const float* __restrict__ proj,
                            const float* __restrict__ kpts,
                            float* __restrict__ out) {
  const int bb = blockIdx.x, tid = threadIdx.x;
  __shared__ float S[16];      // [0]=sw, [1..3]=sum w*p1, [4..6]=sum w*pk, [7..15]=cov
  if (tid < 16) S[tid] = 0.f;
  __syncthreads();

  const float* p1 = kpts + (size_t)bb * NPTS * 3;
  const float* pk = proj + (size_t)bb * NPTS * 3;
  const float* w_ = corr + (size_t)bb * NPTS;

  float lsw = 0.f, la[3] = {0.f, 0.f, 0.f}, lb[3] = {0.f, 0.f, 0.f};
  for (int n = tid; n < NPTS; n += 256) {
    const float w = w_[n];
    lsw += w;
#pragma unroll
    for (int i = 0; i < 3; ++i) {
      la[i] += w * p1[n * 3 + i];
      lb[i] += w * pk[n * 3 + i];
    }
  }
  atomicAdd(&S[0], lsw);
#pragma unroll
  for (int i = 0; i < 3; ++i) { atomicAdd(&S[1 + i], la[i]); atomicAdd(&S[4 + i], lb[i]); }
  __syncthreads();

  const float inv = 1.0f / (S[0] + 1e-5f);            // wn = w * inv
  const float ca[3] = {S[1] * inv, S[2] * inv, S[3] * inv};
  const float cb[3] = {S[4] * inv, S[5] * inv, S[6] * inv};

  float lc[9];
#pragma unroll
  for (int k = 0; k < 9; ++k) lc[k] = 0.f;
  for (int n = tid; n < NPTS; n += 256) {
    const float wn = w_[n] * inv;
    float ax[3], bx[3];
#pragma unroll
    for (int i = 0; i < 3; ++i) {
      ax[i] = p1[n * 3 + i] - ca[i];
      bx[i] = pk[n * 3 + i] - cb[i];
    }
#pragma unroll
    for (int i = 0; i < 3; ++i)
#pragma unroll
      for (int j = 0; j < 3; ++j) lc[i * 3 + j] += wn * ax[i] * bx[j];
  }
#pragma unroll
  for (int k = 0; k < 9; ++k) atomicAdd(&S[7 + k], lc[k]);
  __syncthreads();

  if (tid == 0) {
    float A[3][3];
    for (int i = 0; i < 3; ++i)
      for (int j = 0; j < 3; ++j) A[i][j] = S[7 + i * 3 + j];

    // M = A^T A; Jacobi eigendecomposition -> V (right singular vectors)
    float Mm[3][3];
    for (int i = 0; i < 3; ++i)
      for (int j = 0; j < 3; ++j) {
        float acc = 0.f;
        for (int k = 0; k < 3; ++k) acc += A[k][i] * A[k][j];
        Mm[i][j] = acc;
      }
    float V[3][3] = {{1.f, 0.f, 0.f}, {0.f, 1.f, 0.f}, {0.f, 0.f, 1.f}};
    const int PP[3] = {0, 0, 1}, QQ[3] = {1, 2, 2};
    for (int sweep = 0; sweep < 16; ++sweep) {
      for (int pi = 0; pi < 3; ++pi) {
        const int p = PP[pi], q = QQ[pi];
        const float apq = Mm[p][q];
        if (fabsf(apq) < 1e-18f) continue;
        const float tau = (Mm[q][q] - Mm[p][p]) / (2.f * apq);
        const float t = (tau >= 0.f ? 1.f : -1.f) / (fabsf(tau) + sqrtf(1.f + tau * tau));
        const float c = rsqrtf(1.f + t * t), sn = t * c;
        for (int k = 0; k < 3; ++k) {
          const float mkp = Mm[k][p], mkq = Mm[k][q];
          Mm[k][p] = c * mkp - sn * mkq;
          Mm[k][q] = sn * mkp + c * mkq;
        }
        for (int k = 0; k < 3; ++k) {
          const float mpk = Mm[p][k], mqk = Mm[q][k];
          Mm[p][k] = c * mpk - sn * mqk;
          Mm[q][k] = sn * mpk + c * mqk;
        }
        for (int k = 0; k < 3; ++k) {
          const float vkp = V[k][p], vkq = V[k][q];
          V[k][p] = c * vkp - sn * vkq;
          V[k][q] = sn * vkp + c * vkq;
        }
      }
    }
    float lam[3] = {Mm[0][0], Mm[1][1], Mm[2][2]};
    for (int i = 0; i < 2; ++i)
      for (int j = i + 1; j < 3; ++j)
        if (lam[j] > lam[i]) {
          float tl = lam[i]; lam[i] = lam[j]; lam[j] = tl;
          for (int k = 0; k < 3; ++k) {
            float tv = V[k][i]; V[k][i] = V[k][j]; V[k][j] = tv;
          }
        }
    // U columns: u_i = A v_i / sigma_i (signs consistent with V)
    float U[3][3];
    for (int c3 = 0; c3 < 3; ++c3) {
      const float sig = sqrtf(fmaxf(lam[c3], 0.f));
      float u[3];
      for (int i = 0; i < 3; ++i)
        u[i] = A[i][0] * V[0][c3] + A[i][1] * V[1][c3] + A[i][2] * V[2][c3];
      if (sig > 1e-12f) {
        const float is = 1.f / sig;
        for (int i = 0; i < 3; ++i) U[i][c3] = u[i] * is;
      } else {
        U[0][c3] = U[1][(c3 + 2) % 3] * U[2][(c3 + 1) % 3] -
                   U[2][(c3 + 2) % 3] * U[1][(c3 + 1) % 3];
        U[1][c3] = U[2][(c3 + 2) % 3] * U[0][(c3 + 1) % 3] -
                   U[0][(c3 + 2) % 3] * U[2][(c3 + 1) % 3];
        U[2][c3] = U[0][(c3 + 2) % 3] * U[1][(c3 + 1) % 3] -
                   U[1][(c3 + 2) % 3] * U[0][(c3 + 1) % 3];
      }
    }
    // rot = V U^T; if det <= 0, negate V column 2 (smallest sigma)
    float R[3][3];
    for (int i = 0; i < 3; ++i)
      for (int j = 0; j < 3; ++j)
        R[i][j] = V[i][0] * U[j][0] + V[i][1] * U[j][1] + V[i][2] * U[j][2];
    if (det3(R) <= 0.f) {
      for (int i = 0; i < 3; ++i) V[i][2] = -V[i][2];
      for (int i = 0; i < 3; ++i)
        for (int j = 0; j < 3; ++j)
          R[i][j] = V[i][0] * U[j][0] + V[i][1] * U[j][1] + V[i][2] * U[j][2];
    }
    float* o = out + (size_t)bb * 12;
    for (int i = 0; i < 3; ++i) {
      for (int j = 0; j < 3; ++j) o[i * 4 + j] = R[i][j];
      o[i * 4 + 3] = -(R[i][0] * ca[0] + R[i][1] * ca[1] + R[i][2] * ca[2]) + cb[i];
    }
  }
}

// ---------------------------------------------------------------------------
extern "C" void kernel_launch(void* const* d_in, const int* in_sizes, int n_in,
                              void* d_out, int out_size, void* d_ws, size_t ws_size,
                              hipStream_t stream) {
  (void)in_sizes; (void)n_in; (void)out_size; (void)ws_size;
  const float* feats   = (const float*)d_in[0];   // (16,256,2048) f32
  const float* kpts    = (const float*)d_in[1];   // (16,2048,3)  f32
  const float* epsilon = (const float*)d_in[2];   // (1,)
  const float* gamma   = (const float*)d_in[3];   // (1,)
  float* out = (float*)d_out;                     // (8,3,4)

  char* ws = (char*)d_ws;
  _Float16* fh = (_Float16*)ws;                   // 16 MB f16 normalized features
  size_t off = (size_t)16 * NPTS * NFEAT * sizeof(_Float16);
  float* avec = (float*)(ws + off); off += (size_t)NBATCH * NPTS * sizeof(float);
  float* bvec = (float*)(ws + off); off += (size_t)NBATCH * NPTS * sizeof(float);
  float* corr = (float*)(ws + off); off += (size_t)NBATCH * NPTS * sizeof(float);
  float* proj = (float*)(ws + off);

  k_normalize<<<dim3(NPTS / 256, 16), 256, 0, stream>>>(feats, fh);
  k_init_a<<<(NBATCH * NPTS) / 256, 256, 0, stream>>>(avec);
  for (int it = 0; it < 5; ++it) {
    k_colsum_b<<<dim3(NSLABS, NBATCH), 256, 0, stream>>>(fh, avec, bvec, epsilon, gamma);
    k_rowsum_a<<<dim3(NSLABS, NBATCH), 256, 0, stream>>>(fh, bvec, avec, epsilon, gamma);
  }
  k_final_stats<<<dim3(NSLABS, NBATCH), 256, 0, stream>>>(fh, avec, bvec, kpts,
                                                          corr, proj, epsilon);
  k_transform<<<NBATCH, 256, 0, stream>>>(corr, proj, kpts, out);
}
